// GlobCondWaveNet_71579924955644
// MI455X (gfx1250) — matrix operand order repaired and checked by reading, not compiled, not run on hardware
//
#include <hip/hip_runtime.h>

typedef _Float16 v8h  __attribute__((ext_vector_type(8)));
typedef _Float16 v16h __attribute__((ext_vector_type(16)));
typedef float    v8f  __attribute__((ext_vector_type(8)));

#define B_  4
#define T_  8192
#define C_  128
#define DC_ 128
#define SC_ 256
#define E_  64
#define M_  10
#define L_  20
#define BT_ (B_*T_)
#define MT_ 4                 // time sub-tiles per workgroup in layer kernel (64 timesteps)
#define LP_ 136               // LDS row pitch (f16) for 128-ch tile, 16B aligned
#define HP_ 264               // LDS row pitch (f16) for 256-ch tile, 16B aligned

static __device__ __forceinline__ v16h cat8(v8h lo, v8h hi) {
  return __builtin_shufflevector(lo, hi, 0,1,2,3,4,5,6,7,8,9,10,11,12,13,14,15);
}
// D = A*B + C, 16x16x32 f16 -> f32
static __device__ __forceinline__ v8f wmma_f16(v16h a, v16h b, v8f c) {
  return __builtin_amdgcn_wmma_f32_16x16x32_f16(false, a, false, b, (short)0, c, false, false);
}
// A fragment: row = lane&15; lanes 0-15: K kb+0..7 (v0-3) & kb+16..23 (v4-7); lanes 16-31: +8
static __device__ __forceinline__ v16h ldA(const _Float16* row, int kb, int sub) {
  const _Float16* p = row + kb + sub*8;
  v8h lo = *(const v8h*)(p);
  v8h hi = *(const v8h*)(p + 16);
  return cat8(lo, hi);
}
// B fragment from N-major (row=n, K contiguous): lanes 0-15: K kb+0..7 & kb+16..23; lanes 16-31: +8
static __device__ __forceinline__ v16h ldB(const _Float16* row, int kb, int sub) {
  const _Float16* p = row + kb + sub*8;
  v8h lo = *(const v8h*)(p);
  v8h hi = *(const v8h*)(p + 16);
  return cat8(lo, hi);
}

// ---- prep: transpose f32 [mat][K][N] -> f16 [mat][Npad][K], zero-pad n>=N ----
__global__ void wn_wtrans_k(const float* __restrict__ src, _Float16* __restrict__ dst,
                            int K, int N, int Npad, int total) {
  int id = blockIdx.x * 256 + threadIdx.x;
  if (id >= total) return;
  int k   = id % K;
  int nm  = id / K;
  int n   = nm % Npad;
  int mat = nm / Npad;
  float v = (n < N) ? src[((long)mat * K + k) * N + n] : 0.f;
  dst[id] = (_Float16)v;
}

// ---- prep: condv[l][b][n] = cond[b]@W_cond[l] + b_cond[l] + b_dil[l] ----
__global__ void wn_cond_k(const float* __restrict__ cond, const float* __restrict__ Wc,
                          const float* __restrict__ bc, const float* __restrict__ bd,
                          float* __restrict__ cv) {
  int id = blockIdx.x * 256 + threadIdx.x;
  if (id >= L_ * B_ * SC_) return;
  int n = id % SC_;
  int b = (id / SC_) % B_;
  int l = id / (SC_ * B_);
  float s = bc[l*SC_ + n] + bd[l*SC_ + n];
  for (int e = 0; e < E_; ++e)
    s += cond[b*E_ + e] * Wc[((long)l*E_ + e)*SC_ + n];
  cv[((long)l*B_ + b)*SC_ + n] = s;
}

// ---- prep: x -> x32, x16(buf0); zero agg + zero-pad row ---- (grid covers BT*SC)
__global__ void wn_init_k(const float* __restrict__ x, float* __restrict__ x32,
                          _Float16* __restrict__ x16, float* __restrict__ agg,
                          _Float16* __restrict__ zrow) {
  long id = (long)blockIdx.x * 256 + threadIdx.x;
  if (id < (long)BT_ * C_) {
    float v = x[id];
    x32[id] = v;
    x16[id] = (_Float16)v;
  }
  if (id < C_) zrow[id] = (_Float16)0.f;
  agg[id] = 0.f;
}

// ---- one WaveNet layer: dilated conv + cond + gate + skip + residual ----
// 64 timesteps per workgroup: B fragments loaded once per k-step, reused 4x.
__global__ __launch_bounds__(256) void wn_layer_k(
    const _Float16* __restrict__ xr, _Float16* __restrict__ xw,
    float* __restrict__ x32, float* __restrict__ agg,
    const _Float16* __restrict__ Wd,   // [2][256][128] f16, N-major
    const _Float16* __restrict__ Ws,   // [256][128]
    const _Float16* __restrict__ Wr,   // [128][128]
    const float* __restrict__ cv,      // [B][256] (b_dil+b_cond folded)
    const float* __restrict__ bs, const float* __restrict__ br,
    const _Float16* __restrict__ zrow, int dil)
{
  __shared__ _Float16 lout[MT_*16 * LP_];  // gated output tile, 64 t x 128 ch (padded)
  const int wave = threadIdx.x >> 5;
  const int lane = threadIdx.x & 31;
  const int l15  = lane & 15;
  const int sub  = lane >> 4;
  const int b    = blockIdx.x >> 7;        // T/64 = 128 tiles per batch
  const int t0   = (blockIdx.x & 127) << 6;
  const long bt0 = (long)b * T_ + t0;

  { // ---- stage 1: z = conv(x) + cond; out = tanh(f)*sigmoid(g) ----
    const _Float16* x1p[MT_];
    const _Float16* x0p[MT_];
    #pragma unroll
    for (int mt = 0; mt < MT_; ++mt) {
      const int t  = t0 + mt*16 + l15;
      const int tm = t - dil;
      x1p[mt] = xr + ((long)b * T_ + t) * C_;
      x0p[mt] = (tm >= 0) ? xr + ((long)b * T_ + tm) * C_ : zrow;  // branchless causal pad
    }
    const _Float16* bf0p = Wd + (0*SC_ +       wave*16 + l15) * C_;
    const _Float16* bg0p = Wd + (0*SC_ + 128 + wave*16 + l15) * C_;
    const _Float16* bf1p = Wd + (1*SC_ +       wave*16 + l15) * C_;
    const _Float16* bg1p = Wd + (1*SC_ + 128 + wave*16 + l15) * C_;
    v8f accf[MT_] = {}; v8f accg[MT_] = {};
    #pragma unroll
    for (int kt = 0; kt < 4; ++kt) {
      const int kb = kt * 32;
      const v16h bf0 = ldB(bf0p, kb, sub);
      const v16h bf1 = ldB(bf1p, kb, sub);
      const v16h bg0 = ldB(bg0p, kb, sub);
      const v16h bg1 = ldB(bg1p, kb, sub);
      #pragma unroll
      for (int mt = 0; mt < MT_; ++mt) {
        const v16h a0 = ldA(x0p[mt], kb, sub);
        const v16h a1 = ldA(x1p[mt], kb, sub);
        accf[mt] = wmma_f16(a0, bf0, accf[mt]);
        accg[mt] = wmma_f16(a0, bg0, accg[mt]);
        accf[mt] = wmma_f16(a1, bf1, accf[mt]);
        accg[mt] = wmma_f16(a1, bg1, accg[mt]);
      }
    }
    const float cf = cv[b*SC_ +       wave*16 + l15];
    const float cg = cv[b*SC_ + 128 + wave*16 + l15];
    #pragma unroll
    for (int mt = 0; mt < MT_; ++mt) {
      #pragma unroll
      for (int j = 0; j < 8; ++j) {
        float f = accf[mt][j] + cf;
        float g = accg[mt][j] + cg;
        float o = tanhf(f) * (1.f / (1.f + __expf(-g)));
        lout[(mt*16 + sub*8 + j) * LP_ + wave*16 + l15] = (_Float16)o;
      }
    }
  }
  __syncthreads();
  { // ---- stage 2: skip (2 tiles/wave) + residual (1 tile/wave) ----
    const _Float16* wsk0 = Ws + (wave*32      + l15) * DC_;
    const _Float16* wsk1 = Ws + (wave*32 + 16 + l15) * DC_;
    const _Float16* wre  = Wr + (wave*16      + l15) * DC_;
    v8f as0[MT_] = {}, as1[MT_] = {}, arr[MT_] = {};
    #pragma unroll
    for (int kt = 0; kt < 4; ++kt) {
      const int kb = kt * 32;
      const v16h b0 = ldB(wsk0, kb, sub);
      const v16h b1 = ldB(wsk1, kb, sub);
      const v16h b2 = ldB(wre,  kb, sub);
      #pragma unroll
      for (int mt = 0; mt < MT_; ++mt) {
        const v16h a = ldA(&lout[(mt*16 + l15) * LP_], kb, sub);
        as0[mt] = wmma_f16(a, b0, as0[mt]);
        as1[mt] = wmma_f16(a, b1, as1[mt]);
        arr[mt] = wmma_f16(a, b2, arr[mt]);
      }
    }
    { // skip accumulation (f32, L2 resident)
      const int n0 = wave*32 + l15;
      const int n1 = n0 + 16;
      const float bi0 = bs[n0], bi1 = bs[n1];
      #pragma unroll
      for (int mt = 0; mt < MT_; ++mt) {
        #pragma unroll
        for (int j = 0; j < 8; ++j) {
          const long r = (bt0 + mt*16 + sub*8 + j) * SC_;
          agg[r + n0] += as0[mt][j] + bi0;
          agg[r + n1] += as1[mt][j] + bi1;
        }
      }
    }
    { // residual update: f32 master + f16 copy for next layer
      const int n = wave*16 + l15;
      const float bi = br[n];
      #pragma unroll
      for (int mt = 0; mt < MT_; ++mt) {
        #pragma unroll
        for (int j = 0; j < 8; ++j) {
          const long r = (bt0 + mt*16 + sub*8 + j) * C_ + n;
          float nv = x32[r] + arr[mt][j] + bi;
          x32[r] = nv;
          xw[r]  = (_Float16)nv;
        }
      }
    }
  }
}

// ---- head: relu(agg) -> relu(@W_pre+b) -> @W_fin+b, split into 3 outputs ----
// 32 timesteps per workgroup (2 M sub-tiles).
__global__ __launch_bounds__(256) void wn_final_k(
    const float* __restrict__ agg,
    const _Float16* __restrict__ Wp, const float* __restrict__ bp,
    const _Float16* __restrict__ Wf, const float* __restrict__ bfin,
    float* __restrict__ out)
{
  __shared__ _Float16 lh [32 * HP_];
  __shared__ _Float16 lh2[32 * HP_];
  const int wave = threadIdx.x >> 5;
  const int lane = threadIdx.x & 31;
  const int l15  = lane & 15;
  const int sub  = lane >> 4;
  const long bt0 = (long)blockIdx.x * 32;

  { // load agg tile (32x256), relu -> f16 LDS
    const int r  = threadIdx.x >> 3;        // 0..31
    const int c0 = (threadIdx.x & 7) * 32;  // col chunk
    const float* src = agg + (bt0 + r) * SC_ + c0;
    #pragma unroll
    for (int j = 0; j < 32; ++j) {
      float v = src[j];
      lh[r * HP_ + c0 + j] = (_Float16)(v > 0.f ? v : 0.f);
    }
  }
  __syncthreads();
  { // pre: 256 -> 256, relu
    const _Float16* b0p = Wp + (wave*32      + l15) * SC_;
    const _Float16* b1p = Wp + (wave*32 + 16 + l15) * SC_;
    v8f acc0[2] = {}, acc1[2] = {};
    #pragma unroll
    for (int kt = 0; kt < 8; ++kt) {
      const int kb = kt * 32;
      const v16h b0 = ldB(b0p, kb, sub);
      const v16h b1 = ldB(b1p, kb, sub);
      #pragma unroll
      for (int mt = 0; mt < 2; ++mt) {
        const v16h a = ldA(&lh[(mt*16 + l15) * HP_], kb, sub);
        acc0[mt] = wmma_f16(a, b0, acc0[mt]);
        acc1[mt] = wmma_f16(a, b1, acc1[mt]);
      }
    }
    const int n0 = wave*32 + l15, n1 = n0 + 16;
    const float bi0 = bp[n0], bi1 = bp[n1];
    #pragma unroll
    for (int mt = 0; mt < 2; ++mt) {
      #pragma unroll
      for (int j = 0; j < 8; ++j) {
        float v0 = acc0[mt][j] + bi0; v0 = v0 > 0.f ? v0 : 0.f;
        float v1 = acc1[mt][j] + bi1; v1 = v1 > 0.f ? v1 : 0.f;
        lh2[(mt*16 + sub*8 + j) * HP_ + n0] = (_Float16)v0;
        lh2[(mt*16 + sub*8 + j) * HP_ + n1] = (_Float16)v1;
      }
    }
  }
  __syncthreads();
  if (wave < 2) { // fin: 256 -> 30 (padded to 32), waves 0..1
    const _Float16* bbp = Wf + (wave*16 + l15) * SC_;
    v8f acc[2] = {};
    #pragma unroll
    for (int kt = 0; kt < 8; ++kt) {
      const int kb = kt * 32;
      const v16h bb = ldB(bbp, kb, sub);
      #pragma unroll
      for (int mt = 0; mt < 2; ++mt) {
        const v16h a = ldA(&lh2[(mt*16 + l15) * HP_], kb, sub);
        acc[mt] = wmma_f16(a, bb, acc[mt]);
      }
    }
    const int n = wave*16 + l15;
    if (n < 30) {
      const int comp = n / 10, jj = n % 10;
      const float bi = bfin[n];
      float* dst = out + (long)comp * BT_ * M_ + jj;
      #pragma unroll
      for (int mt = 0; mt < 2; ++mt)
        #pragma unroll
        for (int j = 0; j < 8; ++j)
          dst[(bt0 + mt*16 + sub*8 + j) * M_] = acc[mt][j] + bi;
    }
  }
}

extern "C" void kernel_launch(void* const* d_in, const int* in_sizes, int n_in,
                              void* d_out, int out_size, void* d_ws, size_t ws_size,
                              hipStream_t stream) {
  const float* x      = (const float*)d_in[0];
  const float* cond   = (const float*)d_in[1];
  const float* W_dil  = (const float*)d_in[2];
  const float* b_dil  = (const float*)d_in[3];
  const float* W_cond = (const float*)d_in[4];
  const float* b_cond = (const float*)d_in[5];
  const float* W_res  = (const float*)d_in[6];
  const float* b_res  = (const float*)d_in[7];
  const float* W_skip = (const float*)d_in[8];
  const float* b_skip = (const float*)d_in[9];
  const float* W_pre  = (const float*)d_in[10];
  const float* b_pre  = (const float*)d_in[11];
  const float* W_fin  = (const float*)d_in[12];
  const float* b_fin  = (const float*)d_in[13];

  char* ws = (char*)d_ws;
  float*     x32     = (float*)ws;     ws += (size_t)BT_ * C_ * 4;   // 16 MB
  _Float16*  x16a    = (_Float16*)ws;  ws += (size_t)BT_ * C_ * 2;   // 8 MB
  _Float16*  x16b    = (_Float16*)ws;  ws += (size_t)BT_ * C_ * 2;   // 8 MB
  float*     agg     = (float*)ws;     ws += (size_t)BT_ * SC_ * 4;  // 33.5 MB
  float*     cv      = (float*)ws;     ws += (size_t)L_ * B_ * SC_ * 4;
  _Float16*  Wt_dil  = (_Float16*)ws;  ws += (size_t)L_ * 2 * SC_ * C_ * 2;
  _Float16*  Wt_skip = (_Float16*)ws;  ws += (size_t)L_ * SC_ * DC_ * 2;
  _Float16*  Wt_res  = (_Float16*)ws;  ws += (size_t)L_ * C_ * DC_ * 2;
  _Float16*  Wt_pre  = (_Float16*)ws;  ws += (size_t)SC_ * SC_ * 2;
  _Float16*  Wt_fin  = (_Float16*)ws;  ws += (size_t)32 * SC_ * 2;
  _Float16*  zrow    = (_Float16*)ws;  ws += (size_t)C_ * 2;         // zero row for causal pad

  // weight transposes to f16 N-major
  wn_wtrans_k<<<(1310720 + 255) / 256, 256, 0, stream>>>(W_dil,  Wt_dil,  128, 256, 256, 1310720);
  wn_wtrans_k<<<( 655360 + 255) / 256, 256, 0, stream>>>(W_skip, Wt_skip, 128, 256, 256,  655360);
  wn_wtrans_k<<<( 327680 + 255) / 256, 256, 0, stream>>>(W_res,  Wt_res,  128, 128, 128,  327680);
  wn_wtrans_k<<<(  65536 + 255) / 256, 256, 0, stream>>>(W_pre,  Wt_pre,  256, 256, 256,   65536);
  wn_wtrans_k<<<(   8192 + 255) / 256, 256, 0, stream>>>(W_fin,  Wt_fin,  256,  30,  32,    8192);
  // conditioning table (folds b_dil + b_cond)
  wn_cond_k<<<(L_ * B_ * SC_ + 255) / 256, 256, 0, stream>>>(cond, W_cond, b_cond, b_dil, cv);
  // init residual stream + zero skip accumulator + zero pad row
  wn_init_k<<<(BT_ * SC_) / 256, 256, 0, stream>>>(x, x32, x16a, agg, zrow);

  for (int l = 0; l < L_; ++l) {
    const int dil = 1 << (l % 10);
    const _Float16* xr = (l & 1) ? x16b : x16a;
    _Float16*       xw = (l & 1) ? x16a : x16b;
    wn_layer_k<<<BT_ / (MT_*16), 256, 0, stream>>>(
        xr, xw, x32, agg,
        Wt_dil  + (size_t)l * 2 * SC_ * C_,
        Wt_skip + (size_t)l * SC_ * DC_,
        Wt_res  + (size_t)l * C_ * DC_,
        cv + (size_t)l * B_ * SC_,
        b_skip + (size_t)l * SC_,
        b_res  + (size_t)l * C_,
        zrow, dil);
  }
  wn_final_k<<<BT_ / 32, 256, 0, stream>>>(agg, Wt_pre, b_pre, Wt_fin, b_fin, (float*)d_out);
}
